// Seq2Seq_64304250356036
// MI455X (gfx1250) — compile-verified, matrix operand order checked by reading, not run on hardware
//
#include <hip/hip_runtime.h>

// ---------------------------------------------------------------------------
// Seq2Seq GRU + attention for MI455X (gfx1250, wave32, WMMA).
// Dominant cost: 63 x [64,1792]x[1792,32000] output projection (~462 GFLOP).
// - all weights converted to bf16 once (outW: 114.7MB -> L2-resident)
// - every GEMM runs through v_wmma_f32_16x16x32_bf16
// - logits GEMM: 4-wave blocks share the A tile via double-buffered
//   global_load_async_to_lds_b128 staging (ASYNCcnt), B streamed read-once.
// ---------------------------------------------------------------------------

typedef __bf16 bf16_t;
typedef __attribute__((ext_vector_type(16))) __bf16 v16bf;
typedef __attribute__((ext_vector_type(8)))  float  v8f;

#define BB 64
#define SS 64
#define TT 64
#define HH 512
#define EE 256
#define VV 32000
#define H3 1536   // 3*H
#define H2 1024   // 2*H
#define XK 1280   // E + 2H (decoder GRU input)
#define CK 1792   // H + 2H + E (output projection input)

// ---------------------------------------------------------------------------
// WMMA fragment loader, ISA 7.12.2 16-bit A-matrix 16x32 layout (wave32).
// lane&15 selects the M (or N) row; lane>>4 selects which K-half the lane
// owns; VGPR i holds K pair: k = (i>>2)*16 + half*8 + (i&3)*2.
// A is [M,K] row-major; W is [N,K] row-major (i.e. B^T), same packing.
// Consecutive pairs merge into global_load_b128 / ds_load_b128 at -O3.
// ---------------------------------------------------------------------------
__device__ __forceinline__ v16bf load_frag(const bf16_t* __restrict__ row,
                                           int kh, int k0) {
  v16bf f;
#pragma unroll
  for (int i = 0; i < 8; ++i) {
    int k = k0 + ((i >> 2) << 4) + (kh << 3) + ((i & 3) << 1);
    f[2 * i]     = row[k];
    f[2 * i + 1] = row[k + 1];
  }
  return f;
}

__device__ __forceinline__ v8f wmma_bf16(v16bf a, v16bf b, v8f c) {
  return __builtin_amdgcn_wmma_f32_16x16x32_bf16(
      false, a, false, b, (short)0, c, false, false);
}

// ---------------------------------------------------------------------------
// 1-tile/wave GEMM for the small latency-bound GEMMs (recurrence steps).
// grid = (N/16, M/16), block = 32 (one wave).
// ---------------------------------------------------------------------------
__global__ void gemm_bf16(const bf16_t* __restrict__ A, int lda,
                          const bf16_t* __restrict__ W, int ldw,
                          const float* __restrict__ bias,
                          float* __restrict__ C, long long ldc,
                          int K, int act) {
  const int lane = threadIdx.x;
  const int kh   = lane >> 4;
  const int r16  = lane & 15;
  const bf16_t* Arow = A + (size_t)(blockIdx.y * 16 + r16) * lda;
  const bf16_t* Wrow = W + (size_t)(blockIdx.x * 16 + r16) * ldw;

  v8f acc = {};
  for (int k0 = 0; k0 < K; k0 += 32) {
    if (k0 + 256 < K) __builtin_prefetch(Wrow + k0 + 256, 0, 3);
    v16bf a = load_frag(Arow, kh, k0);
    v16bf b = load_frag(Wrow, kh, k0);
    acc = wmma_bf16(a, b, acc);
  }

  const int n  = blockIdx.x * 16 + r16;
  const float bv = bias ? bias[n] : 0.0f;
  const int m0 = blockIdx.y * 16 + kh * 8; // C/D layout: vgpr r -> M = r + 8*half
#pragma unroll
  for (int r = 0; r < 8; ++r) {
    float v = acc[r] + bv;
    if (act) v = tanhf(v);
    C[(size_t)(m0 + r) * ldc + n] = v;
  }
}

// ---------------------------------------------------------------------------
// Register-blocked GEMM: one wave computes MT x NT 16x16 tiles.
// grid = (N/(16*NT), M/(16*MT)), block = 32.
// ---------------------------------------------------------------------------
template <int MT, int NT>
__global__ void gemm_bf16_mt(const bf16_t* __restrict__ A, int lda,
                             const bf16_t* __restrict__ W, int ldw,
                             const float* __restrict__ bias,
                             float* __restrict__ C, long long ldc,
                             int K, int act) {
  const int lane = threadIdx.x;
  const int kh   = lane >> 4;
  const int r16  = lane & 15;

  const bf16_t* Arow[MT];
#pragma unroll
  for (int mi = 0; mi < MT; ++mi)
    Arow[mi] = A + (size_t)(blockIdx.y * (16 * MT) + mi * 16 + r16) * lda;
  const bf16_t* Wrow[NT];
#pragma unroll
  for (int j = 0; j < NT; ++j)
    Wrow[j] = W + (size_t)(blockIdx.x * (16 * NT) + j * 16 + r16) * ldw;

  v8f acc[MT][NT];
#pragma unroll
  for (int mi = 0; mi < MT; ++mi)
#pragma unroll
    for (int j = 0; j < NT; ++j) acc[mi][j] = (v8f){};

  for (int k0 = 0; k0 < K; k0 += 32) {
    if ((k0 & 127) == 0 && k0 + 512 < K) {
#pragma unroll
      for (int j = 0; j < NT; ++j) __builtin_prefetch(Wrow[j] + k0 + 512, 0, 3);
    }
    v16bf b[NT];
#pragma unroll
    for (int j = 0; j < NT; ++j) b[j] = load_frag(Wrow[j], kh, k0);
#pragma unroll
    for (int mi = 0; mi < MT; ++mi) {
      v16bf a = load_frag(Arow[mi], kh, k0);
#pragma unroll
      for (int j = 0; j < NT; ++j) acc[mi][j] = wmma_bf16(a, b[j], acc[mi][j]);
    }
  }

#pragma unroll
  for (int j = 0; j < NT; ++j) {
    const int n = blockIdx.x * (16 * NT) + j * 16 + r16;
    const float bv = bias ? bias[n] : 0.0f;
#pragma unroll
    for (int mi = 0; mi < MT; ++mi) {
      const int m0 = blockIdx.y * (16 * MT) + mi * 16 + kh * 8;
#pragma unroll
      for (int r = 0; r < 8; ++r) {
        float v = acc[mi][j][r] + bv;
        if (act) v = tanhf(v);
        C[(size_t)(m0 + r) * ldc + n] = v;
      }
    }
  }
}

// ---------------------------------------------------------------------------
// Logits GEMM: C[64, N] = A[64,K] @ W[N,K]^T + bias.
// Block = 128 threads (4 waves); each wave owns 4 N-tiles (block N = 256),
// every wave covers all of M=64 (MT=4) -> W is streamed from L2 exactly once.
// The shared A k-slab (64x32 bf16 = 4KB) is staged into LDS with
// double-buffered GLOBAL_LOAD_ASYNC_TO_LDS_B128 (ASYNCcnt), so per-wave A
// reads come from LDS instead of 4x-replicated L2 traffic.
// grid = (N/256), block = 128.
// ---------------------------------------------------------------------------
__global__ void gemm_logits(const bf16_t* __restrict__ A,   // [64][K]
                            const bf16_t* __restrict__ W,   // [N][K]
                            const float* __restrict__ bias,
                            float* __restrict__ C, long long ldc, int K) {
  __shared__ bf16_t As[2][64 * 32];   // 2 x 4KB double buffer

  const int tid  = threadIdx.x;
  const int lane = tid & 31;
  const int wave = tid >> 5;
  const int kh   = lane >> 4;
  const int r16  = lane & 15;
  const int nbase = blockIdx.x * 256 + wave * 64;

  const bf16_t* Wrow[4];
#pragma unroll
  for (int j = 0; j < 4; ++j)
    Wrow[j] = W + (size_t)(nbase + j * 16 + r16) * K;

  // Stage one 64x32 A-slab into LDS buffer `buf` with async b128 copies.
  // 256 16B-chunks per slab; 128 threads -> 2 chunks each.
  auto stage = [&](int buf, int k0) {
    const unsigned lbase = (unsigned)(unsigned long long)(const void*)&As[buf][0];
    for (int c = tid; c < 256; c += 128) {
      int row = c >> 2, cp = c & 3;
      const bf16_t* g = A + (size_t)row * K + k0 + cp * 8;
      unsigned l = lbase + (unsigned)(row * 32 + cp * 8) * 2;
      asm volatile("global_load_async_to_lds_b128 %0, %1, off"
                   :: "v"(l), "v"(g) : "memory");
    }
  };

  v8f acc[4][4];
#pragma unroll
  for (int mi = 0; mi < 4; ++mi)
#pragma unroll
    for (int j = 0; j < 4; ++j) acc[mi][j] = (v8f){};

  stage(0, 0);
  asm volatile("s_wait_asynccnt 0x0" ::: "memory");
  __syncthreads();

  int buf = 0;
  for (int k0 = 0; k0 < K; k0 += 32) {
    const bool more = (k0 + 32) < K;
    if (more) stage(buf ^ 1, k0 + 32);          // overlap copy with compute

    if ((k0 & 127) == 0 && k0 + 512 < K) {
#pragma unroll
      for (int j = 0; j < 4; ++j) __builtin_prefetch(Wrow[j] + k0 + 512, 0, 3);
    }

    v16bf b[4];
#pragma unroll
    for (int j = 0; j < 4; ++j) b[j] = load_frag(Wrow[j], kh, k0);

    const bf16_t* Abase = &As[buf][0];
#pragma unroll
    for (int mi = 0; mi < 4; ++mi) {
      v16bf a = load_frag(Abase + (mi * 16 + r16) * 32, kh, 0);  // ds_load
#pragma unroll
      for (int j = 0; j < 4; ++j) acc[mi][j] = wmma_bf16(a, b[j], acc[mi][j]);
    }

    if (more) {
      asm volatile("s_wait_asynccnt 0x0" ::: "memory");
      __syncthreads();
      buf ^= 1;
    }
  }

#pragma unroll
  for (int j = 0; j < 4; ++j) {
    const int n = nbase + j * 16 + r16;
    const float bv = bias ? bias[n] : 0.0f;
#pragma unroll
    for (int mi = 0; mi < 4; ++mi) {
      const int m0 = mi * 16 + kh * 8;
#pragma unroll
      for (int r = 0; r < 8; ++r)
        C[(size_t)(m0 + r) * ldc + n] = acc[mi][j][r] + bv;
    }
  }
}

// ---------------------------------------------------------------------------
// Utility kernels
// ---------------------------------------------------------------------------
__global__ void cvt_f32_bf16(const float* __restrict__ src, int src_ld, int cols,
                             bf16_t* __restrict__ dst, int n) {
  int i = blockIdx.x * blockDim.x + threadIdx.x;
  if (i >= n) return;
  int r = i / cols, c = i - r * cols;
  dst[i] = (bf16_t)src[(size_t)r * src_ld + c];
}

__global__ void fill0_f32(float* p, int n) {
  int i = blockIdx.x * blockDim.x + threadIdx.x;
  if (i < n) p[i] = 0.0f;
}
__global__ void fill0_bf16(bf16_t* p, int n) {
  int i = blockIdx.x * blockDim.x + threadIdx.x;
  if (i < n) p[i] = (bf16_t)0.0f;
}
// outputs[:, 0, :] = 0
__global__ void zero_out_t0(float* out) {
  int i = blockIdx.x * blockDim.x + threadIdx.x;
  if (i >= BB * VV) return;
  int b = i / VV, v = i - b * VV;
  out[(size_t)b * TT * VV + v] = 0.0f;
}

// Gather src embeddings into [s][b][E] bf16 so the gi GEMM result is [s][b][3H].
__global__ void gather_src_emb(const int* __restrict__ src,
                               const float* __restrict__ table,
                               bf16_t* __restrict__ out) {
  int i = blockIdx.x * blockDim.x + threadIdx.x;
  if (i >= BB * SS * EE) return;
  int e = i % EE, sb = i / EE;
  int b = sb % BB, s = sb / BB;
  int tok = src[b * SS + s];
  out[i] = (bf16_t)table[(size_t)tok * EE + e];
}

// GRU gates: h2 = (1-z)*n + z*h; writes f32 h, bf16 h (for next GEMM),
// and optional f32/bf16 copies into enc_out / cat buffers.
__global__ void gru_cell(const float* __restrict__ gi, const float* __restrict__ gh,
                         const float* __restrict__ h, float* __restrict__ hout,
                         bf16_t* __restrict__ hbf, int hbf_ld,
                         float* __restrict__ eo, int eo_ld,
                         bf16_t* __restrict__ eobf, int eobf_ld) {
  int i = blockIdx.x * blockDim.x + threadIdx.x;
  if (i >= BB * HH) return;
  int b = i >> 9, j = i & (HH - 1);
  const float* gib = gi + (size_t)b * H3;
  const float* ghb = gh + (size_t)b * H3;
  float r = 1.0f / (1.0f + expf(-(gib[j] + ghb[j])));
  float z = 1.0f / (1.0f + expf(-(gib[HH + j] + ghb[HH + j])));
  float n = tanhf(gib[2 * HH + j] + r * ghb[2 * HH + j]);
  float h2 = (1.0f - z) * n + z * h[i];
  hout[i] = h2;
  hbf[(size_t)b * hbf_ld + j] = (bf16_t)h2;
  if (eo)   eo[(size_t)b * eo_ld + j] = h2;
  if (eobf) eobf[(size_t)b * eobf_ld + j] = (bf16_t)h2;
}

// scores[b][s] = sum_h tanh(hW[b][h] + enc_proj[b][s][h]) * v[h]
// one wave per (b,s); enc_proj already contains attn_b (GEMM bias).
__global__ void attn_scores(const float* __restrict__ hW,
                            const float* __restrict__ encp,
                            const float* __restrict__ v,
                            float* __restrict__ scores) {
  int bs = blockIdx.x;
  int b = bs >> 6;  // S = 64
  int lane = threadIdx.x;
  const float* hp = hW + (size_t)b * HH;
  const float* ep = encp + (size_t)bs * HH;
  float acc = 0.0f;
  for (int h = lane; h < HH; h += 32) acc += tanhf(hp[h] + ep[h]) * v[h];
#pragma unroll
  for (int o = 16; o > 0; o >>= 1) acc += __shfl_xor(acc, o, 32);
  if (lane == 0) scores[bs] = acc;
}

// softmax over S, ctx = aw @ enc_out, and build bf16 GEMM inputs:
//   x_bf  [b][1280] = [emb_t | ctx]          (decoder GRU input)
//   cat_bf[b][1792] = [h2 | ctx | emb_t]     (h2 filled later by gru_cell)
__global__ void softmax_ctx(const float* __restrict__ scores,
                            const float* __restrict__ enc_out,
                            const int* __restrict__ trg, int t,
                            const float* __restrict__ demb,
                            bf16_t* __restrict__ x_bf,
                            bf16_t* __restrict__ cat_bf) {
  int b = blockIdx.x, tid = threadIdx.x; // 256 threads
  __shared__ float aw[SS];
  if (tid < SS) aw[tid] = scores[b * SS + tid];
  __syncthreads();
  if (tid == 0) {
    float m = -1e30f;
    for (int s = 0; s < SS; ++s) m = fmaxf(m, aw[s]);
    float sum = 0.0f;
    for (int s = 0; s < SS; ++s) { aw[s] = expf(aw[s] - m); sum += aw[s]; }
    float inv = 1.0f / sum;
    for (int s = 0; s < SS; ++s) aw[s] *= inv;
  }
  __syncthreads();
  for (int d = tid; d < H2; d += 256) {
    float acc = 0.0f;
    for (int s = 0; s < SS; ++s)
      acc += aw[s] * enc_out[((size_t)b * SS + s) * H2 + d];
    bf16_t v = (bf16_t)acc;
    x_bf[(size_t)b * XK + EE + d] = v;       // ctx after emb
    cat_bf[(size_t)b * CK + HH + d] = v;     // ctx after h2
  }
  int tok = trg[b * TT + t];
  for (int e = tid; e < EE; e += 256) {
    bf16_t v = (bf16_t)demb[(size_t)tok * EE + e];
    x_bf[(size_t)b * XK + e] = v;
    cat_bf[(size_t)b * CK + HH + H2 + e] = v;
  }
}

// ---------------------------------------------------------------------------
// Host driver
// ---------------------------------------------------------------------------
static inline void* wsa(void* base, size_t& off, size_t bytes) {
  void* p = (char*)base + off;
  off += (bytes + 255) & ~(size_t)255;
  return p;
}

#define GEMM(Aptr, lda, Wptr, ldw, biasPtr, Cptr, ldc, M, N, Kk, act)          \
  gemm_bf16<<<dim3((N) / 16, (M) / 16), dim3(32), 0, stream>>>(                \
      (const bf16_t*)(Aptr), (lda), (const bf16_t*)(Wptr), (ldw),             \
      (biasPtr), (Cptr), (long long)(ldc), (Kk), (act))

#define GEMM_MT(MT, NT, Aptr, lda, Wptr, ldw, biasPtr, Cptr, ldc, M, N, Kk, act)\
  gemm_bf16_mt<MT, NT><<<dim3((N) / (16 * (NT)), (M) / (16 * (MT))),           \
                         dim3(32), 0, stream>>>(                               \
      (const bf16_t*)(Aptr), (lda), (const bf16_t*)(Wptr), (ldw),             \
      (biasPtr), (Cptr), (long long)(ldc), (Kk), (act))

#define CVT(srcp, sld, cols, dstp, n)                                          \
  cvt_f32_bf16<<<((n) + 255) / 256, 256, 0, stream>>>((srcp), (sld), (cols),   \
                                                      (dstp), (n))

extern "C" void kernel_launch(void* const* d_in, const int* in_sizes, int n_in,
                              void* d_out, int out_size, void* d_ws,
                              size_t ws_size, hipStream_t stream) {
  const int*   src       = (const int*)d_in[0];
  const int*   trg       = (const int*)d_in[1];
  const float* enc_emb   = (const float*)d_in[2];
  const float* enc_Wih_f = (const float*)d_in[3];
  const float* enc_Whh_f = (const float*)d_in[4];
  const float* enc_bih_f = (const float*)d_in[5];
  const float* enc_bhh_f = (const float*)d_in[6];
  const float* enc_Wih_b = (const float*)d_in[7];
  const float* enc_Whh_b = (const float*)d_in[8];
  const float* enc_bih_b = (const float*)d_in[9];
  const float* enc_bhh_b = (const float*)d_in[10];
  const float* enc_fcW   = (const float*)d_in[11];
  const float* enc_fcb   = (const float*)d_in[12];
  const float* attn_W    = (const float*)d_in[13];
  const float* attn_b    = (const float*)d_in[14];
  const float* attn_v    = (const float*)d_in[15];
  const float* dec_emb   = (const float*)d_in[16];
  const float* dec_Wih   = (const float*)d_in[17];
  const float* dec_Whh   = (const float*)d_in[18];
  const float* dec_bih   = (const float*)d_in[19];
  const float* dec_bhh   = (const float*)d_in[20];
  const float* outW      = (const float*)d_in[21];
  const float* outb      = (const float*)d_in[22];
  float* out = (float*)d_out;

  // ---- workspace layout (bump allocator) ----
  size_t off = 0;
  bf16_t* wWihF   = (bf16_t*)wsa(d_ws, off, (size_t)H3 * EE * 2);
  bf16_t* wWihB   = (bf16_t*)wsa(d_ws, off, (size_t)H3 * EE * 2);
  bf16_t* wHHf    = (bf16_t*)wsa(d_ws, off, (size_t)H3 * HH * 2);
  bf16_t* wHHb    = (bf16_t*)wsa(d_ws, off, (size_t)H3 * HH * 2);
  bf16_t* wFc     = (bf16_t*)wsa(d_ws, off, (size_t)HH * H2 * 2);
  bf16_t* wWaH    = (bf16_t*)wsa(d_ws, off, (size_t)HH * HH * 2);
  bf16_t* wWaE    = (bf16_t*)wsa(d_ws, off, (size_t)HH * H2 * 2);
  bf16_t* wDWih   = (bf16_t*)wsa(d_ws, off, (size_t)H3 * XK * 2);
  bf16_t* wDWhh   = (bf16_t*)wsa(d_ws, off, (size_t)H3 * HH * 2);
  bf16_t* wOutW   = (bf16_t*)wsa(d_ws, off, (size_t)VV * CK * 2);   // 114.7 MB, L2-resident
  bf16_t* emb_sb  = (bf16_t*)wsa(d_ws, off, (size_t)SS * BB * EE * 2);
  float*  gi_f    = (float*)wsa(d_ws, off, (size_t)SS * BB * H3 * 4);
  float*  gi_b    = (float*)wsa(d_ws, off, (size_t)SS * BB * H3 * 4);
  float*  h_f     = (float*)wsa(d_ws, off, (size_t)BB * HH * 4);
  float*  h_b     = (float*)wsa(d_ws, off, (size_t)BB * HH * 4);
  bf16_t* hcatbf  = (bf16_t*)wsa(d_ws, off, (size_t)BB * H2 * 2);   // [hf|hb]
  float*  gh_f    = (float*)wsa(d_ws, off, (size_t)BB * H3 * 4);
  float*  gh_b    = (float*)wsa(d_ws, off, (size_t)BB * H3 * 4);
  float*  enc_out = (float*)wsa(d_ws, off, (size_t)BB * SS * H2 * 4);
  bf16_t* enc_obf = (bf16_t*)wsa(d_ws, off, (size_t)BB * SS * H2 * 2);
  float*  enc_prj = (float*)wsa(d_ws, off, (size_t)BB * SS * HH * 4);
  float*  dec_h   = (float*)wsa(d_ws, off, (size_t)BB * HH * 4);
  bf16_t* dec_hbf = (bf16_t*)wsa(d_ws, off, (size_t)BB * HH * 2);
  float*  hW      = (float*)wsa(d_ws, off, (size_t)BB * HH * 4);
  float*  scores  = (float*)wsa(d_ws, off, (size_t)BB * SS * 4);
  bf16_t* x_bf    = (bf16_t*)wsa(d_ws, off, (size_t)BB * XK * 2);
  bf16_t* cat_bf  = (bf16_t*)wsa(d_ws, off, (size_t)BB * CK * 2);
  float*  giD     = (float*)wsa(d_ws, off, (size_t)BB * H3 * 4);
  float*  ghD     = (float*)wsa(d_ws, off, (size_t)BB * H3 * 4);
  (void)ws_size; (void)in_sizes; (void)n_in; (void)out_size;

  // ---- init ----
  fill0_f32<<<(BB * HH + 255) / 256, 256, 0, stream>>>(h_f, BB * HH);
  fill0_f32<<<(BB * HH + 255) / 256, 256, 0, stream>>>(h_b, BB * HH);
  fill0_bf16<<<(BB * H2 + 255) / 256, 256, 0, stream>>>(hcatbf, BB * H2);
  zero_out_t0<<<(BB * VV + 255) / 256, 256, 0, stream>>>(out);

  // ---- weight conversions f32 -> bf16 (deterministic, every call) ----
  CVT(enc_Wih_f, EE, EE, wWihF, H3 * EE);
  CVT(enc_Wih_b, EE, EE, wWihB, H3 * EE);
  CVT(enc_Whh_f, HH, HH, wHHf, H3 * HH);
  CVT(enc_Whh_b, HH, HH, wHHb, H3 * HH);
  CVT(enc_fcW, H2, H2, wFc, HH * H2);
  CVT(attn_W, H3, HH, wWaH, HH * HH);          // attn_W[:, :H]
  CVT(attn_W + HH, H3, H2, wWaE, HH * H2);     // attn_W[:, H:]
  CVT(dec_Wih, XK, XK, wDWih, H3 * XK);
  CVT(dec_Whh, HH, HH, wDWhh, H3 * HH);
  CVT(outW, CK, CK, wOutW, VV * CK);

  // ---- encoder ----
  gather_src_emb<<<(BB * SS * EE + 255) / 256, 256, 0, stream>>>(src, enc_emb, emb_sb);
  // gi = emb @ Wih^T + bih  (both directions), output layout [s][b][3H]
  GEMM_MT(2, 4, emb_sb, EE, wWihF, EE, enc_bih_f, gi_f, H3, SS * BB, H3, EE, 0);
  GEMM_MT(2, 4, emb_sb, EE, wWihB, EE, enc_bih_b, gi_b, H3, SS * BB, H3, EE, 0);

  const int cellBlocks = (BB * HH + 255) / 256;
  for (int s = 0; s < SS; ++s) {  // forward scan
    GEMM(hcatbf, H2, wHHf, HH, enc_bhh_f, gh_f, H3, BB, H3, HH, 0);
    gru_cell<<<cellBlocks, 256, 0, stream>>>(
        gi_f + (size_t)s * BB * H3, gh_f, h_f, h_f, hcatbf, H2,
        enc_out + (size_t)s * H2, SS * H2,
        enc_obf + (size_t)s * H2, SS * H2);
  }
  for (int s = SS - 1; s >= 0; --s) {  // backward scan
    GEMM(hcatbf + HH, H2, wHHb, HH, enc_bhh_b, gh_b, H3, BB, H3, HH, 0);
    gru_cell<<<cellBlocks, 256, 0, stream>>>(
        gi_b + (size_t)s * BB * H3, gh_b, h_b, h_b, hcatbf + HH, H2,
        enc_out + (size_t)s * H2 + HH, SS * H2,
        enc_obf + (size_t)s * H2 + HH, SS * H2);
  }

  // hidden = tanh([hf|hb] @ fcW^T + fcb)
  GEMM(hcatbf, H2, wFc, H2, enc_fcb, dec_h, HH, BB, HH, H2, 1);
  CVT(dec_h, HH, HH, dec_hbf, BB * HH);

  // enc_proj = enc_out @ Wa_e^T + attn_b  (bias folded here)
  GEMM_MT(2, 4, enc_obf, H2, wWaE, H2, attn_b, enc_prj, HH, BB * SS, HH, H2, 0);

  // ---- decoder: 63 steps ----
  for (int t = 0; t < TT - 1; ++t) {
    GEMM(dec_hbf, HH, wWaH, HH, (const float*)nullptr, hW, HH, BB, HH, HH, 0);
    attn_scores<<<BB * SS, 32, 0, stream>>>(hW, enc_prj, attn_v, scores);
    softmax_ctx<<<BB, 256, 0, stream>>>(scores, enc_out, trg, t, dec_emb,
                                        x_bf, cat_bf);
    GEMM(x_bf, XK, wDWih, XK, dec_bih, giD, H3, BB, H3, XK, 0);
    GEMM(dec_hbf, HH, wDWhh, HH, dec_bhh, ghD, H3, BB, H3, HH, 0);
    gru_cell<<<cellBlocks, 256, 0, stream>>>(giD, ghD, dec_h, dec_h,
                                             dec_hbf, HH, (float*)nullptr, 0,
                                             cat_bf, CK);
    // logits -> out[:, t+1, :]: async-LDS staged A, W streamed once from L2.
    gemm_logits<<<dim3(VV / 256), dim3(128), 0, stream>>>(
        cat_bf, wOutW, outb, out + (size_t)(t + 1) * VV, (long long)TT * VV, CK);
  }
}
